// LeNet5_KAN_37220186587780
// MI455X (gfx1250) — matrix operand order repaired and checked by reading, compiled, not run on hardware
//
#include <hip/hip_runtime.h>
#include <hip/hip_bf16.h>

typedef __attribute__((ext_vector_type(16))) _Float16 v16h;
typedef __attribute__((ext_vector_type(8)))  float    v8f;

// ---------------------------------------------------------------------------
// Cubic B-spline feature: dst[0] = relu(x), dst[1..8] = Cox-de Boor bases.
// Knots t_j = -1 + (j-3)*0.4, j=0..11 (uniform, h=0.4, order 3).
// ---------------------------------------------------------------------------
__device__ inline void eval_feat9(float x, _Float16* dst) {
    float b[11];
#pragma unroll
    for (int j = 0; j < 11; ++j) {
        float t0 = -1.0f + (float)(j - 3) * 0.4f;
        float t1 = -1.0f + (float)(j - 2) * 0.4f;
        b[j] = (x >= t0 && x < t1) ? 1.0f : 0.0f;
    }
#pragma unroll
    for (int k = 1; k <= 3; ++k) {
        float inv = 1.0f / (0.4f * (float)k);
#pragma unroll
        for (int j = 0; j < 11 - 1; ++j) {
            if (j < 11 - k) {
                float tj   = -1.0f + (float)(j - 3) * 0.4f;      // t[j]
                float tjk1 = -1.0f + (float)(j + k - 2) * 0.4f;  // t[j+k+1]
                float left  = (x - tj)   * inv;
                float right = (tjk1 - x) * inv;
                b[j] = left * b[j] + right * b[j + 1];
            }
        }
    }
    dst[0] = (_Float16)(x > 0.0f ? x : 0.0f);
#pragma unroll
    for (int g = 0; g < 8; ++g) dst[1 + g] = (_Float16)b[g];
}

// A-fragment K index for 16-bit 16x32 A layout (half-pair v, lane half)
__device__ inline int a_kidx(int v, int hlf) {
    return (v < 4) ? (2 * v + 8 * hlf) : (16 + 2 * (v - 4) + 8 * hlf);
}

// ---------------------------------------------------------------------------
// Weight packing kernels (f32 -> padded f16, row-per-output [Opad][Kpad])
// ---------------------------------------------------------------------------
__global__ void pack_conv_w(const float* __restrict__ basew,
                            const float* __restrict__ splinew,
                            _Float16* __restrict__ dst,
                            int O, int I, int Opad, int Kpad) {
    int idx = blockIdx.x * blockDim.x + threadIdx.x;
    if (idx >= Opad * Kpad) return;
    int o = idx / Kpad, k = idx % Kpad;
    float v = 0.0f;
    if (o < O) {
        int i = k / 9, t = k % 9;
        if (i < I) v = (t == 0) ? basew[o * I + i]
                                : splinew[(o * I + i) * 8 + (t - 1)];
    }
    dst[idx] = (_Float16)v;
}

__global__ void pack_fc_w(const float* __restrict__ src, _Float16* __restrict__ dst,
                          int O, int K, int Opad, int Kpad) {
    int idx = blockIdx.x * blockDim.x + threadIdx.x;
    if (idx >= Opad * Kpad) return;
    int o = idx / Kpad, k = idx % Kpad;
    dst[idx] = (_Float16)((o < O && k < K) ? src[o * K + k] : 0.0f);
}

__global__ void pack_vec(const float* __restrict__ src, float* __restrict__ dst,
                         int n, int npad) {
    int i = blockIdx.x * blockDim.x + threadIdx.x;
    if (i < npad) dst[i] = (i < n) ? src[i] : 0.0f;
}

// ---------------------------------------------------------------------------
// KAN conv1: x(1024,1,28,28) -> c1(1024,6,24,24).  GEMM (N x 256)@(256 x 16)
// with features [relu,8 bases] per input, cols i*9+t, pad 225..255 = 0.
// 4 waves / block, one 16-patch WMMA tile per wave.
// ---------------------------------------------------------------------------
__global__ __launch_bounds__(128) void kan_conv1(const float* __restrict__ x,
                                                 const _Float16* __restrict__ W,
                                                 float* __restrict__ c1) {
    __shared__ _Float16 feat[4][16][256];   // 32 KB
    const int wave = threadIdx.x >> 5, lane = threadIdx.x & 31;
    const int m = lane & 15, hlf = lane >> 4;
    const int tile = blockIdx.x * 4 + wave;

    // ---- feature fill: lane (m,hlf) covers inputs [hlf?13:0 .. hlf?25:13)
    {
        int n = tile * 16 + m;
        int bimg = n / 576, rem = n % 576, py = rem / 24, px = rem % 24;
        const float* img = x + (size_t)bimg * 784;
        _Float16* frow = feat[wave][m];
        int i0 = hlf ? 13 : 0, i1 = hlf ? 25 : 13;
        for (int i = i0; i < i1; ++i) {
            int dy = i / 5, dx = i % 5;
            float p = img[(py + dy) * 28 + (px + dx)];
            eval_feat9(p, frow + i * 9);
        }
        if (hlf) {
            for (int k = 225; k < 256; ++k) frow[k] = (_Float16)0.0f;
        }
    }
    __syncthreads();

    // ---- 8 WMMA k-steps, K = 256
    v8f acc = {};
    const _Float16* wcol = W + (size_t)m * 256;   // B col = lane%16
    for (int kb = 0; kb < 256; kb += 32) {
        v16h a, b;
#pragma unroll
        for (int v = 0; v < 8; ++v) {
            int ka = kb + a_kidx(v, hlf);
            a[2 * v]     = feat[wave][m][ka];
            a[2 * v + 1] = feat[wave][m][ka + 1];
            int kw = kb + a_kidx(v, hlf);
            b[2 * v]     = wcol[kw];
            b[2 * v + 1] = wcol[kw + 1];
        }
        acc = __builtin_amdgcn_wmma_f32_16x16x32_f16(false, a, false, b,
                                                     (short)0, acc, false, false);
    }

    // ---- scatter store: D lane col = m, VGPR r -> row 8*hlf+r
    if (m < 6) {
#pragma unroll
        for (int r = 0; r < 8; ++r) {
            int nn = tile * 16 + 8 * hlf + r;
            int bb = nn / 576, rr = nn % 576, yy = rr / 24, xx = rr % 24;
            c1[(((size_t)bb * 6 + m) * 24 + yy) * 24 + xx] = acc[r];
        }
    }
}

// ---------------------------------------------------------------------------
// KAN conv2: h1(1024,6,12,12) -> c2(1024,16,8,8).  GEMM (N x 1440)@(1440 x 16)
// K streamed in 5 chunks of 288 cols (32 inputs x 9 features).
// ---------------------------------------------------------------------------
__global__ __launch_bounds__(128) void kan_conv2(const float* __restrict__ h1,
                                                 const _Float16* __restrict__ W,
                                                 float* __restrict__ c2) {
    __shared__ _Float16 feat[4][16][288];   // 36 KB
    const int wave = threadIdx.x >> 5, lane = threadIdx.x & 31;
    const int m = lane & 15, hlf = lane >> 4;
    const int tile = blockIdx.x * 4 + wave;

    const int n = tile * 16 + m;
    const int bimg = n >> 6, rem = n & 63, oy = rem >> 3, ox = rem & 7;
    const float* base = h1 + (size_t)bimg * 6 * 144;
    _Float16* frow = feat[wave][m];

    v8f acc = {};
    const _Float16* wcol = W + (size_t)m * 1440;

    for (int c = 0; c < 5; ++c) {
        __syncthreads();   // previous chunk's LDS reads done before overwrite
        for (int ii = 0; ii < 16; ++ii) {
            int i = c * 32 + hlf * 16 + ii;
            _Float16* d = frow + (hlf * 16 + ii) * 9;
            if (i < 150) {
                int ch = i / 25, idx = i % 25, dy = idx / 5, dx = idx % 5;
                float p = base[(size_t)ch * 144 + (oy + dy) * 12 + (ox + dx)];
                eval_feat9(p, d);
            } else {
#pragma unroll
                for (int t = 0; t < 9; ++t) d[t] = (_Float16)0.0f;
            }
        }
        __syncthreads();

        for (int ks = 0; ks < 9; ++ks) {
            int kb = ks * 32;
            v16h a, b;
#pragma unroll
            for (int v = 0; v < 8; ++v) {
                int ka = kb + a_kidx(v, hlf);
                a[2 * v]     = feat[wave][m][ka];
                a[2 * v + 1] = feat[wave][m][ka + 1];
                int kw = c * 288 + kb + a_kidx(v, hlf);
                b[2 * v]     = wcol[kw];
                b[2 * v + 1] = wcol[kw + 1];
            }
            acc = __builtin_amdgcn_wmma_f32_16x16x32_f16(false, a, false, b,
                                                         (short)0, acc, false, false);
        }
    }

#pragma unroll
    for (int r = 0; r < 8; ++r) {
        int nn = tile * 16 + 8 * hlf + r;
        int bb = nn >> 6, rr = nn & 63, yy = rr >> 3, xx = rr & 7;
        c2[(((size_t)bb * 16 + m) * 8 + yy) * 8 + xx] = acc[r];
    }
}

// ---------------------------------------------------------------------------
// 2x2 average pool (B,C,H,W) -> (B,C,H/2,W/2)
// ---------------------------------------------------------------------------
__global__ void avgpool2_k(const float* __restrict__ in, float* __restrict__ out,
                           int H, int W, int total) {
    int idx = blockIdx.x * blockDim.x + threadIdx.x;
    if (idx >= total) return;
    int Ho = H >> 1, Wo = W >> 1;
    int xo = idx % Wo, yo = (idx / Wo) % Ho, bc = idx / (Wo * Ho);
    const float* p = in + (size_t)bc * H * W + (size_t)(2 * yo) * W + 2 * xo;
    out[idx] = 0.25f * (p[0] + p[1] + p[W] + p[W + 1]);
}

// ---------------------------------------------------------------------------
// WMMA FC: out(row,o) = act(A(row,:)@W(o,:) + bias[o]).  One wave per 16x16
// tile.  A is f32 [N][Ka] (Ka multiple of 32, no masking needed); W is f16
// [Opad][Ka]; out stride Ostride, cols < Ostore written.
// ---------------------------------------------------------------------------
__global__ __launch_bounds__(32) void fc_wmma(const float* __restrict__ A,
                                              const _Float16* __restrict__ W,
                                              const float* __restrict__ bias,
                                              float* __restrict__ out,
                                              int Ka, int Ostride, int Ostore,
                                              int relu) {
    const int lane = threadIdx.x & 31;
    const int m = lane & 15, hlf = lane >> 4;
    const int o = blockIdx.y * 16 + m;
    const float* arow = A + (size_t)(blockIdx.x * 16 + m) * Ka;
    const _Float16* wcol = W + (size_t)o * Ka;

    v8f acc = {};
    for (int kb = 0; kb < Ka; kb += 32) {
        v16h a, b;
#pragma unroll
        for (int v = 0; v < 8; ++v) {
            int ka = kb + a_kidx(v, hlf);
            a[2 * v]     = (_Float16)arow[ka];
            a[2 * v + 1] = (_Float16)arow[ka + 1];
            int kw = kb + a_kidx(v, hlf);
            b[2 * v]     = wcol[kw];
            b[2 * v + 1] = wcol[kw + 1];
        }
        acc = __builtin_amdgcn_wmma_f32_16x16x32_f16(false, a, false, b,
                                                     (short)0, acc, false, false);
    }

    if (o < Ostore) {
        float bv = bias[o];
#pragma unroll
        for (int r = 0; r < 8; ++r) {
            int row = blockIdx.x * 16 + 8 * hlf + r;
            float v = acc[r] + bv;
            if (relu) v = v > 0.0f ? v : 0.0f;
            out[(size_t)row * Ostride + o] = v;
        }
    }
}

// ---------------------------------------------------------------------------
// Host-side orchestration
// ---------------------------------------------------------------------------
static inline size_t align256(size_t x) { return (x + 255) & ~(size_t)255; }

extern "C" void kernel_launch(void* const* d_in, const int* in_sizes, int n_in,
                              void* d_out, int out_size, void* d_ws, size_t ws_size,
                              hipStream_t stream) {
    (void)in_sizes; (void)n_in; (void)out_size; (void)ws_size;
    const float* x         = (const float*)d_in[0];
    const float* base_w1   = (const float*)d_in[1];
    const float* spline_w1 = (const float*)d_in[2];
    const float* base_w2   = (const float*)d_in[3];
    const float* spline_w2 = (const float*)d_in[4];
    const float* fc1_w     = (const float*)d_in[5];
    const float* fc1_b     = (const float*)d_in[6];
    const float* fc2_w     = (const float*)d_in[7];
    const float* fc2_b     = (const float*)d_in[8];
    const float* fc3_w     = (const float*)d_in[9];
    const float* fc3_b     = (const float*)d_in[10];
    float* outp = (float*)d_out;

    // workspace carve-up (~24 MB)
    char* p = (char*)d_ws;
    size_t off = 0;
    auto grab = [&](size_t bytes) { void* r = p + off; off = align256(off + bytes); return r; };
    _Float16* W1  = (_Float16*)grab((size_t)16 * 256 * 2);
    _Float16* W2  = (_Float16*)grab((size_t)16 * 1440 * 2);
    _Float16* Wf1 = (_Float16*)grab((size_t)128 * 256 * 2);
    _Float16* Wf2 = (_Float16*)grab((size_t)96 * 128 * 2);
    _Float16* Wf3 = (_Float16*)grab((size_t)64 * 96 * 2);
    float* b1 = (float*)grab(128 * 4);
    float* b2 = (float*)grab(96 * 4);
    float* b3 = (float*)grab(64 * 4);
    float* c1 = (float*)grab((size_t)1024 * 6 * 24 * 24 * 4);
    float* h1 = (float*)grab((size_t)1024 * 6 * 12 * 12 * 4);
    float* c2 = (float*)grab((size_t)1024 * 16 * 8 * 8 * 4);
    float* h2 = (float*)grab((size_t)1024 * 256 * 4);
    float* r1 = (float*)grab((size_t)1024 * 128 * 4);
    float* r2 = (float*)grab((size_t)1024 * 96 * 4);

    auto cdiv = [](int a, int b) { return (a + b - 1) / b; };

    // weight / bias packing
    pack_conv_w<<<cdiv(16 * 256, 256), 256, 0, stream>>>(base_w1, spline_w1, W1, 6, 25, 16, 256);
    pack_conv_w<<<cdiv(16 * 1440, 256), 256, 0, stream>>>(base_w2, spline_w2, W2, 16, 150, 16, 1440);
    pack_fc_w<<<cdiv(128 * 256, 256), 256, 0, stream>>>(fc1_w, Wf1, 120, 256, 128, 256);
    pack_fc_w<<<cdiv(96 * 128, 256), 256, 0, stream>>>(fc2_w, Wf2, 84, 120, 96, 128);
    pack_fc_w<<<cdiv(64 * 96, 256), 256, 0, stream>>>(fc3_w, Wf3, 62, 84, 64, 96);
    pack_vec<<<1, 128, 0, stream>>>(fc1_b, b1, 120, 128);
    pack_vec<<<1, 96, 0, stream>>>(fc2_b, b2, 84, 96);
    pack_vec<<<1, 64, 0, stream>>>(fc3_b, b3, 62, 64);

    // conv1: 589824 patches -> 36864 tiles -> 9216 blocks (4 waves each)
    kan_conv1<<<9216, 128, 0, stream>>>(x, W1, c1);
    avgpool2_k<<<cdiv(1024 * 6 * 144, 256), 256, 0, stream>>>(c1, h1, 24, 24, 1024 * 6 * 144);

    // conv2: 65536 patches -> 4096 tiles -> 1024 blocks
    kan_conv2<<<1024, 128, 0, stream>>>(h1, W2, c2);
    avgpool2_k<<<cdiv(1024 * 16 * 16, 256), 256, 0, stream>>>(c2, h2, 8, 8, 1024 * 16 * 16);

    // FC chain (N=1024 rows -> 64 row tiles)
    fc_wmma<<<dim3(64, 8), 32, 0, stream>>>(h2, Wf1, b1, r1, 256, 128, 128, 1);
    fc_wmma<<<dim3(64, 6), 32, 0, stream>>>(r1, Wf2, b2, r2, 128, 96, 96, 1);
    fc_wmma<<<dim3(64, 4), 32, 0, stream>>>(r2, Wf3, b3, outp, 96, 62, 62, 0);
}
